// ZambaBlock_66271345377686
// MI455X (gfx1250) — matrix operand order repaired and checked by reading, not compiled, not run on hardware
//
#include <hip/hip_runtime.h>
#include <hip/hip_bf16.h>

// ---------------------------------------------------------------------------
// CDNA5 (gfx1250) Zamba block: all GEMMs via v_wmma_f32_16x16x32_f16.
// Staging: batched global_load_b128 (saddr + 32-bit voffset) -> v_cvt_pk_f16_f32
// -> ds_store_b64, loads issued before any wait so they overlap.
// ---------------------------------------------------------------------------

typedef __attribute__((ext_vector_type(16))) _Float16 v16h;
typedef __attribute__((ext_vector_type(8)))  _Float16 v8h;
typedef __attribute__((ext_vector_type(4)))  _Float16 v4h;
typedef __attribute__((ext_vector_type(8)))  float    v8f;

namespace {
constexpr int Sq  = 1024;   // sequence length
constexpr int Dm  = 512;    // d_model
constexpr int Hh  = 8;      // query heads
constexpr int DHd = 64;     // head dim
constexpr int KVh = 4;      // kv heads
constexpr int DSt = 16;     // d_state
constexpr int DTr = 32;     // dt_rank
constexpr int DCv = 4;      // d_conv
constexpr int DIn = 1024;   // d_inner = 2*Dm
constexpr int DEPTH = 6;
}

__device__ __forceinline__ float siluf(float x) { return x / (1.f + __expf(-x)); }

// ---------------------------------------------------------------------------
// WMMA GEMM:  C[M,N] = act( scale * (A[M,K] x B) + bias )
//   BT==1: B given as [N,K] row-major (weight.T multiply, the common case)
//   BT==0: B given as [K,N] row-major
//   biasMode: 0 none, 1 per-N column bias, 2 per-M row bias
//   act: 0 none, 1 silu, 2 softplus
// 128 threads = 4 wave32; each wave computes a 16x64 slab of a 64x64 tile.
// REQUIRES: M%64==0, N%64==0, K%32==0, lda/ldb multiples of 4, 16B-aligned
// base pointers.  True for every GEMM in this workload.
// ---------------------------------------------------------------------------
template <int BT>
__launch_bounds__(128)
__global__ void gemm_wmma_kernel(const float* __restrict__ A, int lda,
                                 const float* __restrict__ B, int ldb,
                                 const float* __restrict__ bias, int biasMode,
                                 float* __restrict__ C, int ldc,
                                 int K, float scale, int act)
{
    __shared__ __align__(64) _Float16 As[64][32];   // [m][k]
    __shared__ __align__(64) _Float16 Bs[64][32];   // [n][k]  (N-major)

    const int tid  = threadIdx.x;
    const int lane = tid & 31;
    const int wave = tid >> 5;
    const int mBase = blockIdx.y * 64;
    const int nBase = blockIdx.x * 64;

    // staging coordinates: 8 threads per row x 4 floats, 16 rows per pass
    const int sr = tid >> 3;          // 0..15
    const int sc = (tid & 7) * 4;     // 0..28 step 4
    // BT==0 staging: 16 threads per k-row x 4 floats along n, 8 k-rows/pass
    const int bk = tid >> 4;          // 0..7
    const int bn = (tid & 15) * 4;    // 0..60 step 4

    // 32-bit element offsets (saddr + voffset addressing), advanced per K-step
    const int oA = 16 * lda;
    int aOff = (mBase + sr) * lda + sc;
    const int oB    = BT ? 16 * ldb : 8 * ldb;
    const int bStep = BT ? 32 : 32 * ldb;
    int bOff = BT ? (nBase + sr) * ldb + sc : bk * ldb + (nBase + bn);

    v8f acc[4];
#pragma unroll
    for (int f = 0; f < 4; ++f)
#pragma unroll
        for (int r = 0; r < 8; ++r) acc[f][r] = 0.f;

    for (int k0 = 0; k0 < K; k0 += 32) {
        // ---- phase 1: issue all 8 tile loads (no wait in between) ----
        float4 av[4], bv[4];
#pragma unroll
        for (int g = 0; g < 4; ++g) av[g] = *(const float4*)(A + (aOff + g * oA));
#pragma unroll
        for (int g = 0; g < 4; ++g) bv[g] = *(const float4*)(B + (bOff + g * oB));
        aOff += 32;
        bOff += bStep;
        // prefetch next K tile (gfx1250 global_prefetch_b8); offsets already advanced
        if (k0 + 32 < K) {
            __builtin_prefetch(A + aOff, 0, 0);
            __builtin_prefetch(B + bOff, 0, 0);
        }
        // ---- phase 2: convert + store to LDS ----
#pragma unroll
        for (int g = 0; g < 4; ++g) {
            v4h ah;
            ah[0] = (_Float16)av[g].x; ah[1] = (_Float16)av[g].y;
            ah[2] = (_Float16)av[g].z; ah[3] = (_Float16)av[g].w;
            *(v4h*)&As[g * 16 + sr][sc] = ah;
        }
        if (BT) {
#pragma unroll
            for (int g = 0; g < 4; ++g) {
                v4h bh;
                bh[0] = (_Float16)bv[g].x; bh[1] = (_Float16)bv[g].y;
                bh[2] = (_Float16)bv[g].z; bh[3] = (_Float16)bv[g].w;
                *(v4h*)&Bs[g * 16 + sr][sc] = bh;
            }
        } else {
#pragma unroll
            for (int g = 0; g < 4; ++g) {
                int kk = g * 8 + bk;
                Bs[bn + 0][kk] = (_Float16)bv[g].x;
                Bs[bn + 1][kk] = (_Float16)bv[g].y;
                Bs[bn + 2][kk] = (_Float16)bv[g].z;
                Bs[bn + 3][kk] = (_Float16)bv[g].w;
            }
        }
        __syncthreads();

        // A fragment: lanes 0-15 hold K0-15, lanes 16-31 hold K16-31 (matches B)
        const int rowA = wave * 16 + (lane & 15);
        const int kbA  = (lane >> 4) * 16;
        v8h alo = *(const v8h*)&As[rowA][kbA];
        v8h ahi = *(const v8h*)&As[rowA][kbA + 8];
        v16h a;
#pragma unroll
        for (int i = 0; i < 8; ++i) { a[i] = alo[i]; a[8 + i] = ahi[i]; }

        const int nB  = lane & 15;
        const int kbB = (lane >> 4) * 16;   // lanes 0-15: K0-15, lanes 16-31: K16-31
#pragma unroll
        for (int f = 0; f < 4; ++f) {
            v16h b = *(const v16h*)&Bs[f * 16 + nB][kbB];
            acc[f] = __builtin_amdgcn_wmma_f32_16x16x32_f16(
                false, a, false, b, (short)0, acc[f], false, false);
        }
        __syncthreads();
    }

    // Epilogue: C layout — VGPR r: lanes 0-15 -> M=r, lanes 16-31 -> M=8+r; N=lane&15
    const int mOff  = mBase + wave * 16 + ((lane >> 4) * 8);
    const int nOff0 = nBase + (lane & 15);
#pragma unroll
    for (int f = 0; f < 4; ++f) {
        int gn = nOff0 + f * 16;
        float bn_ = (biasMode == 1) ? bias[gn] : 0.f;
#pragma unroll
        for (int r = 0; r < 8; ++r) {
            int gm = mOff + r;
            float v = acc[f][r] * scale + bn_;
            if (biasMode == 2) v += bias[gm];
            if (act == 1)      v = v / (1.f + __expf(-v));
            else if (act == 2) v = (v > 20.f) ? v : log1pf(__expf(v));
            C[gm * ldc + gn] = v;
        }
    }
}

// --------------------------- elementwise kernels ---------------------------

__launch_bounds__(256)
__global__ void rmsnorm_kernel(const float* __restrict__ x, const float* __restrict__ w,
                               float* __restrict__ out, int cols)
{
    __shared__ float red[256];
    const float* xr = x + (size_t)blockIdx.x * cols;
    float* orw = out + (size_t)blockIdx.x * cols;
    float s = 0.f;
    for (int c = threadIdx.x; c < cols; c += 256) { float v = xr[c]; s += v * v; }
    red[threadIdx.x] = s; __syncthreads();
    for (int o = 128; o > 0; o >>= 1) {
        if (threadIdx.x < o) red[threadIdx.x] += red[threadIdx.x + o];
        __syncthreads();
    }
    float inv = rsqrtf(red[0] / cols + 1e-5f);
    for (int c = threadIdx.x; c < cols; c += 256) orw[c] = xr[c] * inv * w[c];
}

__launch_bounds__(256)
__global__ void layernorm_kernel(const float* __restrict__ x, const float* __restrict__ w,
                                 const float* __restrict__ b, float* __restrict__ out, int cols)
{
    __shared__ float red[256];
    __shared__ float red2[256];
    const float* xr = x + (size_t)blockIdx.x * cols;
    float* orw = out + (size_t)blockIdx.x * cols;
    float s = 0.f, s2 = 0.f;
    for (int c = threadIdx.x; c < cols; c += 256) { float v = xr[c]; s += v; s2 += v * v; }
    red[threadIdx.x] = s; red2[threadIdx.x] = s2; __syncthreads();
    for (int o = 128; o > 0; o >>= 1) {
        if (threadIdx.x < o) { red[threadIdx.x] += red[threadIdx.x + o];
                               red2[threadIdx.x] += red2[threadIdx.x + o]; }
        __syncthreads();
    }
    float mean = red[0] / cols;
    float var  = red2[0] / cols - mean * mean;
    float inv  = rsqrtf(var + 1e-5f);
    for (int c = threadIdx.x; c < cols; c += 256)
        orw[c] = (xr[c] - mean) * inv * w[c] + b[c];
}

__launch_bounds__(64)
__global__ void l2norm64_kernel(float* __restrict__ x)
{
    __shared__ float red[64];
    float* xr = x + (size_t)blockIdx.x * 64;
    float v = xr[threadIdx.x];
    red[threadIdx.x] = v * v; __syncthreads();
    for (int o = 32; o > 0; o >>= 1) {
        if (threadIdx.x < o) red[threadIdx.x] += red[threadIdx.x + o];
        __syncthreads();
    }
    xr[threadIdx.x] = v * rsqrtf(red[0] + 1e-12f);
}

__launch_bounds__(256)
__global__ void softmax_rows_kernel(float* __restrict__ x, int cols)
{
    __shared__ float red[256];
    float* xr = x + (size_t)blockIdx.x * cols;
    float m = -3.4e38f;
    for (int c = threadIdx.x; c < cols; c += 256) m = fmaxf(m, xr[c]);
    red[threadIdx.x] = m; __syncthreads();
    for (int o = 128; o > 0; o >>= 1) {
        if (threadIdx.x < o) red[threadIdx.x] = fmaxf(red[threadIdx.x], red[threadIdx.x + o]);
        __syncthreads();
    }
    m = red[0]; __syncthreads();
    float s = 0.f;
    for (int c = threadIdx.x; c < cols; c += 256) { float e = __expf(xr[c] - m); xr[c] = e; s += e; }
    red[threadIdx.x] = s; __syncthreads();
    for (int o = 128; o > 0; o >>= 1) {
        if (threadIdx.x < o) red[threadIdx.x] += red[threadIdx.x + o];
        __syncthreads();
    }
    float inv = 1.f / red[0];
    for (int c = threadIdx.x; c < cols; c += 256) xr[c] *= inv;
}

__launch_bounds__(256)
__global__ void conv_silu_kernel(const float* __restrict__ xz, const float* __restrict__ cw,
                                 const float* __restrict__ cb, float* __restrict__ out)
{
    int i = blockIdx.x * 256 + threadIdx.x;
    if (i >= Sq * DIn) return;
    int d = i % DIn, l = i / DIn;
    float acc = cb[d];
#pragma unroll
    for (int j = 0; j < DCv; ++j) {
        int ls = l - (DCv - 1) + j;
        if (ls >= 0) acc += cw[d * DCv + j] * xz[(size_t)ls * (2 * DIn) + d];
    }
    out[i] = siluf(acc);
}

// Selective scan: one thread per channel d; h[16] in registers; fused D-skip
// and SiLU(z) gating.  dbl rows = [dt(32) | B(16) | C(16)].
__launch_bounds__(256)
__global__ void sel_scan_kernel(const float* __restrict__ delta, const float* __restrict__ xic,
                                const float* __restrict__ dbl,   const float* __restrict__ xz,
                                const float* __restrict__ A_log, const float* __restrict__ Dp,
                                float* __restrict__ y)
{
    int d = blockIdx.x * 256 + threadIdx.x;
    if (d >= DIn) return;
    float A[DSt], h[DSt];
#pragma unroll
    for (int s = 0; s < DSt; ++s) { A[s] = -__expf(A_log[d * DSt + s]); h[s] = 0.f; }
    const float Dpd = Dp[d];
    for (int l = 0; l < Sq; ++l) {
        float dlt = delta[(size_t)l * DIn + d];
        float xi  = xic[(size_t)l * DIn + d];
        float z   = xz[(size_t)l * (2 * DIn) + DIn + d];
        const float* bl = dbl + (size_t)l * 64;
        float acc = 0.f;
#pragma unroll
        for (int s = 0; s < DSt; ++s) {
            float dA = __expf(dlt * A[s]);
            h[s] = dA * h[s] + dlt * bl[DTr + s] * xi;
            acc += h[s] * bl[DTr + DSt + s];
        }
        float v = acc + Dpd * xi;
        y[(size_t)l * DIn + d] = v * siluf(z);
    }
}

__launch_bounds__(256)
__global__ void axpby_kernel(const float* __restrict__ a, const float* __restrict__ b,
                             float* __restrict__ out, float alpha, float beta, int n)
{
    int i = blockIdx.x * 256 + threadIdx.x;
    if (i < n) out[i] = alpha * a[i] + beta * b[i];
}

// ------------------------------- host side ---------------------------------

extern "C" void kernel_launch(void* const* d_in, const int* in_sizes, int n_in,
                              void* d_out, int out_size, void* d_ws, size_t ws_size,
                              hipStream_t stream)
{
    (void)in_sizes; (void)n_in; (void)out_size; (void)ws_size;
    auto in = [&](int i) { return (const float*)d_in[i]; };

    struct MambaP {
        const float *norm_w, *in_w, *conv_w, *conv_b, *xproj_w,
                    *dt_w, *dt_b, *A_log, *Dp, *out_w;
    };
    MambaP m1 { in(1), in(2), in(3), in(4), in(5), in(6), in(7), in(8), in(9), in(10) };
    MambaP m2 { in(11), in(12), in(13), in(14), in(15), in(16), in(17), in(18), in(19), in(20) };
    const float *f1_ln_w = in(21), *f1_ln_b = in(22);
    const float *f2_ln_w = in(23), *f2_ln_b = in(24);
    const float *wq = in(25), *wk = in(26), *wv = in(27), *wo = in(28);
    const float *sh_ln_w = in(29), *sh_ln_b = in(30);
    const float *ff_w1 = in(31), *ff_b1 = in(32), *ff_ln_w = in(33), *ff_ln_b = in(34);
    const float *ff_w2 = in(35), *ff_b2 = in(36);
    const float *proj_w = in(37), *proj_b = in(38);
    const float *omni1_w = in(39), *omni1_b = in(40);
    const float *omni2_w = in(41), *omni2_b = in(42);
    const float *out_ln_w = in(43), *out_ln_b = in(44);
    const float *x_in = in(0);

    // ---- workspace carve-up (floats) ----
    float* ws = (float*)d_ws;
    size_t off = 0;
    auto alloc = [&](size_t n) { float* p = ws + off; off += n; return p; };
    float* X      = alloc((size_t)Sq * Dm);
    float* T0     = alloc((size_t)Sq * Dm);
    float* XZ     = alloc((size_t)Sq * 2 * DIn);
    float* XIC    = alloc((size_t)Sq * DIn);
    float* DBL    = alloc((size_t)Sq * 64);
    float* DELTA  = alloc((size_t)Sq * DIn);
    float* Y      = alloc((size_t)Sq * DIn);
    float* T1     = alloc((size_t)Sq * Dm);
    float* FIRST  = alloc((size_t)Sq * Dm);
    float* SECOND = alloc((size_t)Sq * Dm);
    float* CONC   = alloc((size_t)2 * Sq * Dm);
    float* C1     = alloc((size_t)Sq * Dm);
    float* SH     = alloc((size_t)Sq * Dm);
    float* P      = alloc((size_t)Sq * Dm);
    float* Qb     = alloc((size_t)Sq * Hh * DHd);
    float* Kb     = alloc((size_t)Sq * KVh * DHd);
    float* Vb     = alloc((size_t)Sq * KVh * DHd);
    float* LOGI   = alloc((size_t)Sq * Sq);
    float* Ob     = alloc((size_t)Sq * Hh * DHd);
    float* AT     = alloc((size_t)Sq * Dm);
    float* FF     = alloc((size_t)Sq * 4 * Dm);

    // bT: 1 = B is [N,K] (weight.T), 0 = B is [K,N]
    auto gemm = [&](const float* A, int lda, const float* B, int ldb, int bT,
                    const float* bias, int biasMode, float* C, int ldc,
                    int M, int N, int K, float scale, int act) {
        dim3 grid((unsigned)(N / 64), (unsigned)(M / 64));
        if (bT)
            gemm_wmma_kernel<1><<<grid, dim3(128), 0, stream>>>(
                A, lda, B, ldb, bias, biasMode, C, ldc, K, scale, act);
        else
            gemm_wmma_kernel<0><<<grid, dim3(128), 0, stream>>>(
                A, lda, B, ldb, bias, biasMode, C, ldc, K, scale, act);
    };

    auto mixer = [&](float* Xc, const MambaP& p) {
        rmsnorm_kernel<<<Sq, 256, 0, stream>>>(Xc, p.norm_w, T0, Dm);
        // in_proj: [S,2DI]
        gemm(T0, Dm, p.in_w, Dm, 1, nullptr, 0, XZ, 2 * DIn, Sq, 2 * DIn, Dm, 1.f, 0);
        // causal depthwise conv + silu
        conv_silu_kernel<<<(Sq * DIn + 255) / 256, 256, 0, stream>>>(XZ, p.conv_w, p.conv_b, XIC);
        // x_proj: [S, 64]
        gemm(XIC, DIn, p.xproj_w, DIn, 1, nullptr, 0, DBL, 64, Sq, 64, DIn, 1.f, 0);
        // delta = softplus(dt @ dt_w.T + dt_b): [S, DI]
        gemm(DBL, 64, p.dt_w, DTr, 1, p.dt_b, 1, DELTA, DIn, Sq, DIn, DTr, 1.f, 2);
        // selective scan with fused D-skip and silu(z) gate
        sel_scan_kernel<<<(DIn + 255) / 256, 256, 0, stream>>>(
            DELTA, XIC, DBL, XZ, p.A_log, p.Dp, Y);
        // out_proj
        gemm(Y, DIn, p.out_w, DIn, 1, nullptr, 0, T1, Dm, Sq, Dm, DIn, 1.f, 0);
        // x = mixer + 2x
        axpby_kernel<<<(Sq * Dm + 255) / 256, 256, 0, stream>>>(T1, Xc, Xc, 1.f, 2.f, Sq * Dm);
    };

    auto fractal = [&](float* Xc, const MambaP& p, const float* lnw, const float* lnb) {
        for (int it = 0; it < DEPTH; ++it) mixer(Xc, p);
        layernorm_kernel<<<Sq, 256, 0, stream>>>(Xc, lnw, lnb, Xc, Dm);
    };

    auto shared_blk = [&](const float* Xi, float* Out) {
        gemm(Xi, Dm, wq, Dm, 1, nullptr, 0, Qb, Hh * DHd, Sq, Hh * DHd, Dm, 1.f, 0);
        gemm(Xi, Dm, wk, Dm, 1, nullptr, 0, Kb, KVh * DHd, Sq, KVh * DHd, Dm, 1.f, 0);
        gemm(Xi, Dm, wv, Dm, 1, nullptr, 0, Vb, KVh * DHd, Sq, KVh * DHd, Dm, 1.f, 0);
        l2norm64_kernel<<<Sq * Hh, 64, 0, stream>>>(Qb);
        l2norm64_kernel<<<Sq * KVh, 64, 0, stream>>>(Kb);
        for (int h = 0; h < Hh; ++h) {
            const float* qh = Qb + h * DHd;
            const float* kh = Kb + (h / (Hh / KVh)) * DHd;
            const float* vh = Vb + (h / (Hh / KVh)) * DHd;
            // logits = 10 * q k^T
            gemm(qh, Hh * DHd, kh, KVh * DHd, 1, nullptr, 0, LOGI, Sq, Sq, Sq, DHd, 10.f, 0);
            softmax_rows_kernel<<<Sq, 256, 0, stream>>>(LOGI, Sq);
            // o_h = softmax @ v   (B is [K=S, N=64], non-transposed)
            gemm(LOGI, Sq, vh, KVh * DHd, 0, nullptr, 0, Ob + h * DHd, Hh * DHd,
                 Sq, DHd, Sq, 1.f, 0);
        }
        gemm(Ob, Hh * DHd, wo, Hh * DHd, 1, nullptr, 0, AT, Dm, Sq, Dm, Hh * DHd, 1.f, 0);
        layernorm_kernel<<<Sq, 256, 0, stream>>>(AT, sh_ln_w, sh_ln_b, AT, Dm);
        gemm(AT, Dm, ff_w1, Dm, 1, ff_b1, 1, FF, 4 * Dm, Sq, 4 * Dm, Dm, 1.f, 1 /*silu*/);
        layernorm_kernel<<<Sq, 256, 0, stream>>>(FF, ff_ln_w, ff_ln_b, FF, 4 * Dm);
        gemm(FF, 4 * Dm, ff_w2, 4 * Dm, 1, ff_b2, 1, Out, Dm, Sq, Dm, 4 * Dm, 1.f, 0);
        layernorm_kernel<<<Sq, 256, 0, stream>>>(Out, sh_ln_w, sh_ln_b, Out, Dm);
    };

    auto omni = [&](const float* skip, const float* xcur, const float* w, const float* b,
                    float* Out) {
        hipMemcpyAsync(CONC, skip, sizeof(float) * Sq * Dm, hipMemcpyDeviceToDevice, stream);
        hipMemcpyAsync(CONC + (size_t)Sq * Dm, xcur, sizeof(float) * Sq * Dm,
                       hipMemcpyDeviceToDevice, stream);
        // out[t,d] = sum_s w[t,s] * conc[s,d] + b[t]   (bias per-M row)
        gemm(w, 2 * Sq, CONC, Dm, 0, b, 2, Out, Dm, Sq, Dm, 2 * Sq, 1.f, 0);
    };

    const int nSD = Sq * Dm;

    // ---------------- Zamba dataflow ----------------
    hipMemcpyAsync(X, x_in, sizeof(float) * nSD, hipMemcpyDeviceToDevice, stream);
    fractal(X, m1, f1_ln_w, f1_ln_b);
    hipMemcpyAsync(FIRST, X, sizeof(float) * nSD, hipMemcpyDeviceToDevice, stream);

    omni(x_in, X, omni1_w, omni1_b, C1);
    shared_blk(C1, SH);
    gemm(SH, Dm, proj_w, Dm, 1, proj_b, 1, P, Dm, Sq, Dm, Dm, 1.f, 0);
    axpby_kernel<<<(nSD + 255) / 256, 256, 0, stream>>>(P, FIRST, X, 1.f, 1.f, nSD);

    fractal(X, m2, f2_ln_w, f2_ln_b);
    hipMemcpyAsync(SECOND, X, sizeof(float) * nSD, hipMemcpyDeviceToDevice, stream);

    omni(x_in, X, omni2_w, omni2_b, C1);
    shared_blk(C1, SH);
    gemm(SH, Dm, proj_w, Dm, 1, proj_b, 1, P, Dm, Sq, Dm, Dm, 1.f, 0);
    axpby_kernel<<<(nSD + 255) / 256, 256, 0, stream>>>(P, SECOND, X, 1.f, 1.f, nSD);

    fractal(X, m1, f1_ln_w, f1_ln_b);   // fractal reused with m1/f1
    layernorm_kernel<<<Sq, 256, 0, stream>>>(X, out_ln_w, out_ln_b, (float*)d_out, Dm);
}
